// RESpecificRNN_25838523252870
// MI455X (gfx1250) — compile-verified, matrix operand order checked by reading, not run on hardware
//
#include <hip/hip_runtime.h>
#include <hip/hip_bf16.h>
#include <math.h>

// ---------------- problem constants ----------------
#define T_   256
#define B_   128
#define S_   512
#define IN_  1024   // 2*S
#define NER_ 25
#define E_   1049   // 2*S + NER
#define K_   8
#define R_   16
#define HID_ 128
#define W2C_ 273    // 2*HID + 1 + R
#define EP_  1056   // E padded up to multiple of 32 for WMMA K loop

typedef _Float16 f16;
typedef __attribute__((ext_vector_type(16))) _Float16 v16h;
typedef __attribute__((ext_vector_type(8)))  float    v8f;
typedef __attribute__((ext_vector_type(4)))  unsigned int u32x4;
typedef __attribute__((ext_vector_type(8)))  int      i32x8;
typedef __attribute__((ext_vector_type(4)))  int      i32x4;

// ---------------- fragment loaders ----------------
// A-fragment (16x32, 16-bit): lane row = o0 + (lane&15); lanes 0-15 hold
// K in {0..7, 16..23}, lanes 16-31 hold K in {8..15, 24..31}. Two 16B chunks.
// B-fragment (32x16): lane = output column; weights stored row-major W[n][k],
// i.e. B[k][n] = W[n][k], so the same contiguous-along-k loader applies.
__device__ __forceinline__ v16h ldfrag(const f16* __restrict__ p, int ld,
                                       int o0, int k0, int lane) {
  const int o  = o0 + (lane & 15);
  const int c0 = (lane < 16) ? 0 : 8;
  const f16* q = p + (size_t)o * ld + k0 + c0;
  v16h r;
#pragma unroll
  for (int i = 0; i < 8; ++i) { r[i] = q[i]; r[i + 8] = q[16 + i]; }
  return r;
}

// fp32 source with K-bound guard (for W1 remainder fragment), converts to f16
__device__ __forceinline__ v16h ldfrag_f32g(const float* __restrict__ p, int ld,
                                            int o0, int k0, int lane, int kmax) {
  const int o  = o0 + (lane & 15);
  const int c0 = (lane < 16) ? 0 : 8;
  const float* q = p + (size_t)o * ld;
  v16h r;
#pragma unroll
  for (int i = 0; i < 8; ++i) {
    const int ka = k0 + c0 + i;
    const int kb = k0 + c0 + 16 + i;
    r[i]     = (ka < kmax) ? (f16)q[ka] : (f16)0.0f;
    r[i + 8] = (kb < kmax) ? (f16)q[kb] : (f16)0.0f;
  }
  return r;
}

__device__ __forceinline__ v8f wmma_f16(v16h a, v16h b, v8f c) {
  return __builtin_amdgcn_wmma_f32_16x16x32_f16(false, a, false, b,
                                                (short)0, c, false, false);
}

__device__ __forceinline__ float sigf(float x) { return 1.0f / (1.0f + expf(-x)); }

// ---------------- Tensor Data Mover: 2-D tile (global f16 -> LDS) ----------------
// D# group0: count=1 | lds_addr | global_addr[56:0] | type=2
// D# group1: data_size=1(2B); tensor_dim0=rowlen; tensor_dim1=rows;
//            tile_dim0=rowlen; tile_dim1=rows; tensor_dim0_stride=row stride.
#if __has_builtin(__builtin_amdgcn_tensor_load_to_lds)
#define HAVE_TDM 1
__device__ __forceinline__ void tdm_load_2d(const f16* gsrc, f16* ldst,
                                            int rows, int rowlen, int rstride) {
  const unsigned long long ga = (unsigned long long)(uintptr_t)gsrc;
  const unsigned int laddr   = (unsigned int)(uintptr_t)ldst; // low 32b = LDS offset
  u32x4 g0;
  g0[0] = 1u;                                        // count=1 (valid), user mode
  g0[1] = laddr;                                     // lds_addr (bytes)
  g0[2] = (unsigned int)(ga & 0xFFFFFFFFu);          // global_addr[31:0]
  g0[3] = (unsigned int)((ga >> 32) & 0x1FFFFFFu) | (2u << 30); // [56:32] | type=2
  i32x8 g1;
  g1[0] = (1 << 16);                                 // data_size = 1 -> 2 bytes
  g1[1] = (rowlen & 0xFFFF) << 16;                   // tensor_dim0[15:0]
  g1[2] = ((rowlen >> 16) & 0xFFFF) | ((rows & 0xFFFF) << 16); // dim0 hi | dim1 lo
  g1[3] = ((rows >> 16) & 0xFFFF) | ((rowlen & 0xFFFF) << 16); // dim1 hi | tile_dim0
  g1[4] = rows & 0xFFFF;                             // tile_dim1 (tile_dim2 = 0)
  g1[5] = rstride;                                   // tensor_dim0_stride[31:0]
  g1[6] = 0;
  g1[7] = 0;
  const i32x4 z4 = {0, 0, 0, 0};
  const i32x8 z8 = {0, 0, 0, 0, 0, 0, 0, 0};
  // 6-arg form (clang-23 / therock-10.0 headers): (g0, g1, g2, g3, g4, cpol)
  __builtin_amdgcn_tensor_load_to_lds(g0, g1, z4, z4, z8, 0);
}
#else
#define HAVE_TDM 0
#endif

// cooperative fallback / generic stage (all threads of the block)
__device__ __forceinline__ void stage_copy(const f16* __restrict__ gsrc,
                                           f16* __restrict__ ldst,
                                           int rows, int rowlen, int rstride,
                                           int tid, int nthr) {
  for (int x = tid; x < rows * rowlen; x += nthr) {
    const int r = x / rowlen, c = x - r * rowlen;
    ldst[x] = gsrc[(size_t)r * rstride + c];
  }
}

// ---------------- helper kernels ----------------
__global__ void k_f32_to_f16(const float* __restrict__ s, f16* __restrict__ d,
                             size_t n) {
  size_t i = (size_t)blockIdx.x * blockDim.x + threadIdx.x;
  const size_t stride = (size_t)gridDim.x * blockDim.x;
  for (; i < n; i += stride) d[i] = (f16)s[i];
}

__global__ void k_zero_state(f16* __restrict__ hbf, float* __restrict__ hf, int n) {
  int i = blockIdx.x * blockDim.x + threadIdx.x;
  if (i < n) { hbf[i] = (f16)0.0f; hf[i] = 0.0f; }
}

// ---------------- fused GRU step (both directions), WMMA f16 + TDM ----------------
// grid = (64, 2), block = 128 (4 waves). All 4 waves share one 16-row batch tile;
// TDM stages the x_t tile (16x1024 f16) and h_prev tile (16x512 f16) into LDS,
// waves stream the L2-resident weight B-fragments from global and issue WMMAs.
__global__ __launch_bounds__(128) void k_gru_step(
    const f16* __restrict__ Xh,        // [T][B][IN] f16
    const f16* __restrict__ Wih,       // [2][3S][IN] f16
    const f16* __restrict__ Whh,       // [2][3S][S]  f16
    const float* __restrict__ bih_f, const float* __restrict__ bhh_f,
    const float* __restrict__ bih_b, const float* __restrict__ bhh_b,
    f16* __restrict__ Hbf,             // [2][2][B][S] f16 (double-buffered state)
    float* __restrict__ Hf,            // [2][2][B][S] f32
    float* __restrict__ OutF,          // [T][B][S]
    float* __restrict__ OutB,          // [T][B][S]
    int step) {
  __shared__ f16 sX[16 * IN_];   // 32 KB
  __shared__ f16 sHp[16 * S_];   // 16 KB

  const int dir  = blockIdx.y;
  const int t    = dir ? (T_ - 1 - step) : step;
  const int lane = threadIdx.x & 31;
  const int wid  = threadIdx.x >> 5;                 // 0..3
  const int m0   = (blockIdx.x >> 3) * 16;           // batch tile (shared in block)
  const int j0   = ((blockIdx.x & 7) * 4 + wid) * 16; // hidden tile per wave

  const f16* Xt    = Xh  + (size_t)t * B_ * IN_;
  const f16* Wihd  = Wih + (size_t)dir * 3 * S_ * IN_;
  const f16* Whhd  = Whh + (size_t)dir * 3 * S_ * S_;
  const f16* hprev = Hbf + ((size_t)dir * 2 + (step & 1)) * B_ * S_;
  f16*  hnext      = Hbf + ((size_t)dir * 2 + ((step + 1) & 1)) * B_ * S_;
  const float* hpf = Hf  + ((size_t)dir * 2 + (step & 1)) * B_ * S_;
  float* hnf       = Hf  + ((size_t)dir * 2 + ((step + 1) & 1)) * B_ * S_;
  float* outp      = (dir ? OutB : OutF) + (size_t)t * B_ * S_;
  const float* bih = dir ? bih_b : bih_f;
  const float* bhh = dir ? bhh_b : bhh_f;

  // stage activation tiles into LDS via the Tensor Data Mover
#if HAVE_TDM
  if (wid == 0) {
    tdm_load_2d(Xt + (size_t)m0 * IN_,   sX,  16, IN_, IN_);
    tdm_load_2d(hprev + (size_t)m0 * S_, sHp, 16, S_,  S_);
    __builtin_amdgcn_s_wait_tensorcnt(0);
  }
#else
  stage_copy(Xt + (size_t)m0 * IN_,   sX,  16, IN_, IN_, threadIdx.x, 128);
  stage_copy(hprev + (size_t)m0 * S_, sHp, 16, S_,  S_,  threadIdx.x, 128);
#endif
  __syncthreads();

  v8f accI[3], accH[3];
#pragma unroll
  for (int g = 0; g < 3; ++g) { v8f z = {}; accI[g] = z; accH[g] = z; }

  // gi: K = 1024 over x_t (A from LDS, B = Wih from global/L2)
  for (int k0 = 0; k0 < IN_; k0 += 32) {
    __builtin_prefetch(Wihd + (size_t)j0 * IN_ + k0 + 256, 0, 0);
    const v16h a = ldfrag(sX, IN_, 0, k0, lane);
#pragma unroll
    for (int g = 0; g < 3; ++g) {
      const v16h b = ldfrag(Wihd, IN_, j0 + g * S_, k0, lane);
      accI[g] = wmma_f16(a, b, accI[g]);
    }
  }
  // gh: K = 512 over h_prev (A from LDS, B = Whh from global/L2)
  for (int k0 = 0; k0 < S_; k0 += 32) {
    const v16h a = ldfrag(sHp, S_, 0, k0, lane);
#pragma unroll
    for (int g = 0; g < 3; ++g) {
      const v16h b = ldfrag(Whhd, S_, j0 + g * S_, k0, lane);
      accH[g] = wmma_f16(a, b, accH[g]);
    }
  }

  // gate math on the C-fragment layout: col = j0 + lane%16, row = v + 8*(lane>=16)
  const int j  = j0 + (lane & 15);
  const float bi0 = bih[j], bi1 = bih[j + S_], bi2 = bih[j + 2 * S_];
  const float bh0 = bhh[j], bh1 = bhh[j + S_], bh2 = bhh[j + 2 * S_];
#pragma unroll
  for (int v = 0; v < 8; ++v) {
    const int m = m0 + v + ((lane >> 4) << 3);
    const float r = sigf(accI[0][v] + bi0 + accH[0][v] + bh0);
    const float z = sigf(accI[1][v] + bi1 + accH[1][v] + bh1);
    const float n = tanhf(accI[2][v] + bi2 + r * (accH[2][v] + bh2));
    const float hold = hpf[(size_t)m * S_ + j];
    const float hnew = (1.0f - z) * n + z * hold;
    outp[(size_t)m * S_ + j] = hnew;
    hnf[(size_t)m * S_ + j]  = hnew;
    hnext[(size_t)m * S_ + j] = (f16)hnew;
  }
}

// ---------------- per-timestep relation head + loss ----------------
// one block (256 threads = 8 waves) per t
__global__ __launch_bounds__(256) void k_pairs(
    const float* __restrict__ OutF, const float* __restrict__ OutB,
    const float* __restrict__ Ner,                 // [T][B][NER]
    const int* __restrict__ Rs, const int* __restrict__ Re,
    const int* __restrict__ Rr,                    // [T][K]
    const float* __restrict__ Mdiag,               // [R][E]
    const float* __restrict__ W1, const float* __restrict__ b1,
    const float* __restrict__ W2, const float* __restrict__ b2,
    float* __restrict__ Tpart) {
  const int t   = blockIdx.x;
  const int tid = threadIdx.x;

  __shared__ f16   sT[16][EP_];     // token embeddings, f16, zero-padded to 1056
  __shared__ float sH[16][HID_];    // W1-projected tokens
  __shared__ float sN[16];          // norms
  __shared__ float sDM[120 * R_];   // distmult features
  __shared__ float sCos[120];
  __shared__ int   sIdx[16], sRs[K_], sRe[K_], sRr[K_];
  __shared__ unsigned char sII[120], sJJ[120];
  __shared__ float sRed[256];

  if (tid < K_) {
    sRs[tid] = Rs[t * K_ + tid];
    sRe[tid] = Re[t * K_ + tid];
    sRr[tid] = Rr[t * K_ + tid];
  }
  if (tid == 0) {
    int c = 0;
    for (int i = 0; i < 16; ++i)
      for (int j = i + 1; j < 16; ++j) { sII[c] = (unsigned char)i; sJJ[c] = (unsigned char)j; ++c; }
  }
  __syncthreads();
  if (tid < 16) sIdx[tid] = (tid < K_) ? sRs[tid] : sRe[tid - K_];
  __syncthreads();

  // gather 16 tokens: [fwd(512) | bwd(512) | ner(25) | pad(7)] -> f16 LDS
  for (int x = tid; x < 16 * EP_; x += 256) {
    const int e = x / EP_, c = x % EP_;
    const int bb = sIdx[e];
    float v;
    if (c < S_)            v = OutF[((size_t)t * B_ + bb) * S_ + c];
    else if (c < 2 * S_)   v = OutB[((size_t)t * B_ + bb) * S_ + (c - S_)];
    else if (c < E_)       v = Ner[((size_t)t * B_ + bb) * NER_ + (c - 2 * S_)];
    else                   v = 0.0f;
    sT[e][c] = (f16)v;
  }
  __syncthreads();

  // token projection: 16xE @ ExHID via WMMA; wave w -> output cols [16w,16w+16)
  {
    const int wv = tid >> 5, lane = tid & 31;
    v8f acc = {};
    for (int k0 = 0; k0 < EP_; k0 += 32) {
      const v16h a = ldfrag(&sT[0][0], EP_, 0, k0, lane);
      const v16h b = ldfrag_f32g(W1, E_, wv * 16, k0, lane, E_);
      acc = wmma_f16(a, b, acc);
    }
    const int n = wv * 16 + (lane & 15);
#pragma unroll
    for (int v = 0; v < 8; ++v) {
      const int row = v + ((lane >> 4) << 3);
      sH[row][n] = acc[v] + b1[n];
    }
  }
  __syncthreads();

  // norms (16 tokens) and raw pair dots (both only read sH)
  if (tid < 16) {
    float s = 0.0f;
    for (int c = 0; c < HID_; ++c) s += sH[tid][c] * sH[tid][c];
    sN[tid] = fmaxf(sqrtf(s), 1e-8f);
  }
  for (int p = tid; p < 120; p += 256) {
    const int i = sII[p], j = sJJ[p];
    float d = 0.0f;
    for (int c = 0; c < HID_; ++c) d += sH[i][c] * sH[j][c];
    sCos[p] = d;
  }
  __syncthreads();

  for (int p = tid; p < 120; p += 256)
    sCos[p] = sCos[p] / (sN[sII[p]] * sN[sJJ[p]]);

  // diag-DistMult: dm[p][r] = sum_e e1*Md[r]*e2
  for (int task = tid; task < 120 * R_; task += 256) {
    const int p = task >> 4, r = task & 15;
    const int i = sII[p], j = sJJ[p];
    const float* md = Mdiag + (size_t)r * E_;
    float s = 0.0f;
    for (int e = 0; e < E_; ++e)
      s += (float)sT[i][e] * (float)sT[j][e] * md[e];
    sDM[task] = s;
  }
  __syncthreads();

  // logits, sigmoid, threshold, labels, BCE (mean over R, sum over pairs)
  float lsum = 0.0f;
  for (int task = tid; task < 120 * R_; task += 256) {
    const int p = task >> 4, r = task & 15;
    const int i = sII[p], j = sJJ[p];
    const float* w = W2 + (size_t)r * W2C_;
    float acc = b2[r];
    for (int c = 0; c < HID_; ++c) acc += sH[i][c] * w[c];
    for (int c = 0; c < HID_; ++c) acc += sH[j][c] * w[HID_ + c];
    acc += sCos[p] * w[2 * HID_];
#pragma unroll
    for (int r2 = 0; r2 < R_; ++r2) acc += sDM[p * R_ + r2] * w[2 * HID_ + 1 + r2];
    const float pr   = sigf(acc);
    const float pred = (pr > 0.9f) ? pr : 0.0f;
    const int a = sIdx[i], bb = sIdx[j];
    float lab = 0.0f;
#pragma unroll
    for (int k = 0; k < K_; ++k)
      if (sRs[k] == a && sRe[k] == bb && sRr[k] == r) lab = 1.0f;
    const float logp  = (pred > 0.0f) ? fmaxf(logf(pred), -100.0f) : -100.0f;
    const float pm    = fminf(pred, 1.0f - 1e-7f);
    const float l1m   = fmaxf(log1pf(-pm), -100.0f);
    lsum += -(lab * logp + (1.0f - lab) * l1m);
  }
  lsum *= (1.0f / (float)R_);

  sRed[tid] = lsum;
  __syncthreads();
  for (int s = 128; s > 0; s >>= 1) {
    if (tid < s) sRed[tid] += sRed[tid + s];
    __syncthreads();
  }
  if (tid == 0) Tpart[t] = sRed[0];
}

__global__ void k_final(const float* __restrict__ Tpart, float* __restrict__ out) {
  __shared__ float sRed[256];
  const int tid = threadIdx.x;
  sRed[tid] = Tpart[tid];
  __syncthreads();
  for (int s = 128; s > 0; s >>= 1) {
    if (tid < s) sRed[tid] += sRed[tid + s];
    __syncthreads();
  }
  if (tid == 0) out[0] = sRed[0];
}

// ---------------- host-side launch ----------------
extern "C" void kernel_launch(void* const* d_in, const int* in_sizes, int n_in,
                              void* d_out, int out_size, void* d_ws, size_t ws_size,
                              hipStream_t stream) {
  (void)in_sizes; (void)n_in; (void)out_size; (void)ws_size;
  const float* X      = (const float*)d_in[0];   // (T,B,IN)
  const float* Ner    = (const float*)d_in[1];   // (T,B,NER)
  const int*   Rs     = (const int*)d_in[2];     // (T,K)
  const int*   Re     = (const int*)d_in[3];
  const int*   Rr     = (const int*)d_in[4];
  const float* Wih_f  = (const float*)d_in[5];   // (3S,IN)
  const float* Whh_f  = (const float*)d_in[6];   // (3S,S)
  const float* bih_f  = (const float*)d_in[7];
  const float* bhh_f  = (const float*)d_in[8];
  const float* Wih_b  = (const float*)d_in[9];
  const float* Whh_b  = (const float*)d_in[10];
  const float* bih_b  = (const float*)d_in[11];
  const float* bhh_b  = (const float*)d_in[12];
  const float* Mdiag  = (const float*)d_in[13];  // (R,E)
  const float* W1     = (const float*)d_in[14];  // (HID,E)
  const float* b1     = (const float*)d_in[15];
  const float* W2     = (const float*)d_in[16];  // (R,273)
  const float* b2     = (const float*)d_in[17];

  // workspace layout (bytes)
  char* w = (char*)d_ws;
  f16*   Xh   = (f16*)  (w + 0);            // 256*128*1024*2      = 67108864
  f16*   Wih  = (f16*)  (w + 67108864);     // 2*1536*1024*2       =  6291456
  f16*   Whh  = (f16*)  (w + 73400320);     // 2*1536*512*2        =  3145728
  f16*   Hbf  = (f16*)  (w + 76546048);     // 2*2*128*512*2       =   524288
  float* Hf   = (float*)(w + 77070336);     // 2*2*128*512*4       =  1048576
  float* OutF = (float*)(w + 78118912);     // 256*128*512*4       = 67108864
  float* OutB = (float*)(w + 145227776);    // 256*128*512*4       = 67108864
  float* Tp   = (float*)(w + 212336640);    // 256*4

  const size_t nX   = (size_t)T_ * B_ * IN_;
  const size_t nWih = (size_t)3 * S_ * IN_;
  const size_t nWhh = (size_t)3 * S_ * S_;

  // convert inputs/weights to f16
  k_f32_to_f16<<<8192, 256, 0, stream>>>(X, Xh, nX);
  k_f32_to_f16<<<2048, 256, 0, stream>>>(Wih_f, Wih,        nWih);
  k_f32_to_f16<<<2048, 256, 0, stream>>>(Wih_b, Wih + nWih, nWih);
  k_f32_to_f16<<<1024, 256, 0, stream>>>(Whh_f, Whh,        nWhh);
  k_f32_to_f16<<<1024, 256, 0, stream>>>(Whh_b, Whh + nWhh, nWhh);

  // zero GRU state (both dirs, both double-buffers)
  const int nState = 2 * 2 * B_ * S_;
  k_zero_state<<<(nState + 255) / 256, 256, 0, stream>>>(Hbf, Hf, nState);

  // sequential recurrence: one launch per step, both directions per launch
  for (int i = 0; i < T_; ++i) {
    k_gru_step<<<dim3(64, 2), 128, 0, stream>>>(
        Xh, Wih, Whh, bih_f, bhh_f, bih_b, bhh_b,
        Hbf, Hf, OutF, OutB, i);
  }

  // relation head + per-t loss
  k_pairs<<<T_, 256, 0, stream>>>(OutF, OutB, Ner, Rs, Re, Rr,
                                  Mdiag, W1, b1, W2, b2, Tp);

  // final scalar
  k_final<<<1, 256, 0, stream>>>(Tp, (float*)d_out);
}
